// SeparableSelfAttention_83872121356575
// MI455X (gfx1250) — matrix operand order repaired and checked by reading, not compiled, not run on hardware
//
#include <hip/hip_runtime.h>

#define DIM 1024
#define NSEQ 4096
#define NBATCH 8
#define MT 64            // rows per workgroup tile in fused kernel
#define XPAD 1032        // padded LDS row stride (bf16 elems): breaks bank conflicts

typedef __attribute__((ext_vector_type(16))) __bf16 v16bf;
typedef __attribute__((ext_vector_type(8)))  float  v8f;

union Frag16 { v16bf v; uint4 u[2]; };

// ---------------------------------------------------------------------------
// Kernel 1: s[b,n] = dot(x[b,n,:], WI)   (one wave per row)
// ---------------------------------------------------------------------------
__global__ __launch_bounds__(256) void k_gemv_wi(const float* __restrict__ x,
                                                 const float* __restrict__ wi,
                                                 float* __restrict__ s) {
    const int row  = blockIdx.x * 8 + (threadIdx.x >> 5);
    const int lane = threadIdx.x & 31;
    const float* xr = x + (size_t)row * DIM;
    float acc = 0.f;
    #pragma unroll
    for (int i = lane * 4; i < DIM; i += 128) {
        float4 xv = *(const float4*)(xr + i);
        float4 wv = *(const float4*)(wi + i);
        acc += xv.x * wv.x + xv.y * wv.y + xv.z * wv.z + xv.w * wv.w;
    }
    #pragma unroll
    for (int o = 16; o > 0; o >>= 1) acc += __shfl_down(acc, o, 32);
    if (lane == 0) s[row] = acc;
}

// ---------------------------------------------------------------------------
// Kernel 2: softmax over n (per batch) + partial weighted row-sums of x.
// grid (B, 16 n-chunks) x 1024 threads. Writes xpart[b][chunk][d].
// ---------------------------------------------------------------------------
__global__ __launch_bounds__(1024) void k_softmax_xpart(const float* __restrict__ x,
                                                        const float* __restrict__ s,
                                                        float* __restrict__ xpart) {
    const int b  = blockIdx.x;
    const int nc = blockIdx.y;            // 16 chunks of 256 rows
    const int tid  = threadIdx.x;
    const int lane = tid & 31, wid = tid >> 5;
    __shared__ float redm[32];
    __shared__ float reds[32];
    __shared__ float cs[256];
    const float* sb = s + b * NSEQ;

    float m = -3.4e38f;
    for (int i = tid; i < NSEQ; i += 1024) m = fmaxf(m, sb[i]);
    #pragma unroll
    for (int o = 16; o > 0; o >>= 1) m = fmaxf(m, __shfl_down(m, o, 32));
    if (lane == 0) redm[wid] = m;
    __syncthreads();
    if (wid == 0) {
        float t = redm[lane];
        #pragma unroll
        for (int o = 16; o > 0; o >>= 1) t = fmaxf(t, __shfl_down(t, o, 32));
        if (lane == 0) redm[0] = t;
    }
    __syncthreads();
    m = redm[0];

    float sum = 0.f;
    for (int i = tid; i < NSEQ; i += 1024) sum += __expf(sb[i] - m);
    #pragma unroll
    for (int o = 16; o > 0; o >>= 1) sum += __shfl_down(sum, o, 32);
    if (lane == 0) reds[wid] = sum;
    __syncthreads();
    if (wid == 0) {
        float t = reds[lane];
        #pragma unroll
        for (int o = 16; o > 0; o >>= 1) t += __shfl_down(t, o, 32);
        if (lane == 0) reds[0] = t;
    }
    __syncthreads();
    const float inv = 1.0f / reds[0];
    if (tid < 256) cs[tid] = __expf(sb[nc * 256 + tid] - m) * inv;
    __syncthreads();

    // xpart[b][nc][d] = sum over this chunk's 256 rows of cs[n]*x[b,n,d]
    const int d = tid;  // 0..1023
    const float* xb = x + ((size_t)b * NSEQ + nc * 256) * DIM;
    float acc = 0.f;
    #pragma unroll 4
    for (int n = 0; n < 256; ++n) acc += cs[n] * xb[(size_t)n * DIM + d];
    xpart[((size_t)b * 16 + nc) * DIM + d] = acc;
}

// ---------------------------------------------------------------------------
// Kernel 3: reduce partials -> xbar[b][d]
// ---------------------------------------------------------------------------
__global__ __launch_bounds__(256) void k_reduce_xbar(const float* __restrict__ xpart,
                                                     float* __restrict__ xbar) {
    const int i = blockIdx.x * 256 + threadIdx.x;  // i = b*1024 + d, 8192 total
    const int b = i >> 10, d = i & 1023;
    float acc = 0.f;
    #pragma unroll
    for (int p = 0; p < 16; ++p) acc += xpart[((size_t)b * 16 + p) * DIM + d];
    xbar[i] = acc;
}

// ---------------------------------------------------------------------------
// Kernel 4: cv[b][d] = sum_k xbar[b][k] * WK[k][d]   (tiny GEMM, f32)
// ---------------------------------------------------------------------------
__global__ __launch_bounds__(128) void k_cv(const float* __restrict__ xbar,
                                            const float* __restrict__ wk,
                                            float* __restrict__ cv) {
    const int b = blockIdx.y;
    const int d = blockIdx.x * 128 + threadIdx.x;
    const float* xr = xbar + b * DIM;
    float acc = 0.f;
    #pragma unroll 4
    for (int k = 0; k < DIM; ++k) acc += xr[k] * wk[(size_t)k * DIM + d];
    cv[b * DIM + d] = acc;
}

// ---------------------------------------------------------------------------
// Kernel 5: W[k][n] f32 -> Wt[n][k] bf16 (column-major for B-fragment loads)
// ---------------------------------------------------------------------------
__global__ __launch_bounds__(256) void k_transpose_bf16(const float* __restrict__ w,
                                                        __bf16* __restrict__ wt) {
    const int k = blockIdx.x;
    const int n = blockIdx.y * 256 + threadIdx.x;
    wt[(size_t)n * DIM + k] = (__bf16)w[(size_t)k * DIM + n];
}

// ---------------------------------------------------------------------------
// Kernel 6 (main): per 64-row tile:
//   LDS <- bf16(x tile);  z = bf16(relu(x@WVt) * cv[b]) in LDS;  out = z@WOt
// 8 waves; wave w owns output columns [w*128, w*128+128).
// N register-blocked 4x: each A fragment from LDS feeds 4 WMMAs.
// ---------------------------------------------------------------------------
__global__ __launch_bounds__(256) void k_fused(const float* __restrict__ x,
                                               const __bf16* __restrict__ wvt,
                                               const __bf16* __restrict__ wot,
                                               const float* __restrict__ cv,
                                               float* __restrict__ out) {
    extern __shared__ __align__(16) char smem_raw[];
    __bf16* xs = (__bf16*)smem_raw;          // [MT][XPAD]
    __bf16* zs = xs + MT * XPAD;             // [MT][XPAD]

    const int tid  = threadIdx.x;
    const int lane = tid & 31, wave = tid >> 5;
    const int lm = lane & 15, kh = lane >> 4;
    const size_t row0 = (size_t)blockIdx.x * MT;
    const int b = (int)(row0 >> 12);         // row0 / 4096

    // ---- load x tile, convert to bf16, store to padded LDS ----
    for (int i = tid; i < MT * DIM / 4; i += 256) {
        const int e = i * 4;
        const int r = e >> 10, c = e & (DIM - 1);
        float4 v = *(const float4*)(x + (row0 + r) * DIM + c);
        union { __bf16 h[4]; uint2 u; } pk;
        pk.h[0] = (__bf16)v.x; pk.h[1] = (__bf16)v.y;
        pk.h[2] = (__bf16)v.z; pk.h[3] = (__bf16)v.w;
        *(uint2*)(xs + r * XPAD + c) = pk.u;
    }
    __syncthreads();

    const int n0w = wave * 128;

    // ---- GEMM1: z = bf16(relu(x @ WVt) * cv[b]) ----
    for (int ntg = 0; ntg < 2; ++ntg) {
        const int ncol0 = n0w + ntg * 64 + lm;       // + j*16 per sub-tile
        v8f acc[4][4] = {};                          // [mt][j]
        const __bf16* bbase = wvt + (size_t)ncol0 * DIM + kh * 16;
        for (int ks = 0; ks < 32; ++ks) {
            Frag16 bf[4];
            #pragma unroll
            for (int j = 0; j < 4; ++j) {
                const __bf16* bp = bbase + (size_t)(j * 16) * DIM + ks * 32;
                bf[j].u[0] = *(const uint4*)(bp);
                bf[j].u[1] = *(const uint4*)(bp + 8);
            }
            #pragma unroll
            for (int mt = 0; mt < 4; ++mt) {
                Frag16 af;
                const __bf16* ap = xs + (mt * 16 + lm) * XPAD + ks * 32 + kh * 16;
                af.u[0] = *(const uint4*)(ap);
                af.u[1] = *(const uint4*)(ap + 8);
                #pragma unroll
                for (int j = 0; j < 4; ++j)
                    acc[mt][j] = __builtin_amdgcn_wmma_f32_16x16x32_bf16(
                        false, af.v, false, bf[j].v, (short)0, acc[mt][j], false, false);
            }
        }
        #pragma unroll
        for (int j = 0; j < 4; ++j) {
            const int ncol = ncol0 + j * 16;
            const float cvv = cv[b * DIM + ncol];
            #pragma unroll
            for (int mt = 0; mt < 4; ++mt) {
                #pragma unroll
                for (int r = 0; r < 8; ++r) {
                    const int row = mt * 16 + r + 8 * kh;   // documented C/D layout
                    zs[row * XPAD + ncol] = (__bf16)(fmaxf(acc[mt][j][r], 0.0f) * cvv);
                }
            }
        }
    }
    __syncthreads();

    // ---- GEMM2: out = z @ WOt ----
    for (int ntg = 0; ntg < 2; ++ntg) {
        const int ncol0 = n0w + ntg * 64 + lm;
        v8f acc[4][4] = {};
        const __bf16* bbase = wot + (size_t)ncol0 * DIM + kh * 16;
        for (int ks = 0; ks < 32; ++ks) {
            Frag16 bf[4];
            #pragma unroll
            for (int j = 0; j < 4; ++j) {
                const __bf16* bp = bbase + (size_t)(j * 16) * DIM + ks * 32;
                bf[j].u[0] = *(const uint4*)(bp);
                bf[j].u[1] = *(const uint4*)(bp + 8);
            }
            #pragma unroll
            for (int mt = 0; mt < 4; ++mt) {
                Frag16 af;
                const __bf16* ap = zs + (mt * 16 + lm) * XPAD + ks * 32 + kh * 16;
                af.u[0] = *(const uint4*)(ap);
                af.u[1] = *(const uint4*)(ap + 8);
                #pragma unroll
                for (int j = 0; j < 4; ++j)
                    acc[mt][j] = __builtin_amdgcn_wmma_f32_16x16x32_bf16(
                        false, af.v, false, bf[j].v, (short)0, acc[mt][j], false, false);
            }
        }
        // nontemporal stores: out is write-once, keep x/weights resident in L2
        #pragma unroll
        for (int j = 0; j < 4; ++j) {
            const int ncol = ncol0 + j * 16;
            #pragma unroll
            for (int mt = 0; mt < 4; ++mt) {
                #pragma unroll
                for (int r = 0; r < 8; ++r) {
                    const int row = mt * 16 + r + 8 * kh;
                    __builtin_nontemporal_store(acc[mt][j][r],
                                                &out[(row0 + row) * DIM + ncol]);
                }
            }
        }
    }
}

// ---------------------------------------------------------------------------
extern "C" void kernel_launch(void* const* d_in, const int* in_sizes, int n_in,
                              void* d_out, int out_size, void* d_ws, size_t ws_size,
                              hipStream_t stream) {
    const float* x  = (const float*)d_in[0];
    const float* WI = (const float*)d_in[1];
    const float* WK = (const float*)d_in[2];
    const float* WV = (const float*)d_in[3];
    const float* WO = (const float*)d_in[4];
    float* out = (float*)d_out;

    // workspace layout
    float*  s     = (float*)d_ws;                    // 32768
    float*  xpart = s + NBATCH * NSEQ;               // 8*16*1024 = 131072
    float*  xbar  = xpart + NBATCH * 16 * DIM;       // 8192
    float*  cvb   = xbar + NBATCH * DIM;             // 8192
    __bf16* wvt   = (__bf16*)(cvb + NBATCH * DIM);   // 1M bf16
    __bf16* wot   = wvt + (size_t)DIM * DIM;         // 1M bf16

    // weight conversion / transpose (bf16, column-major)
    k_transpose_bf16<<<dim3(DIM, DIM / 256), 256, 0, stream>>>(WV, wvt);
    k_transpose_bf16<<<dim3(DIM, DIM / 256), 256, 0, stream>>>(WO, wot);

    // s = x @ WI
    k_gemv_wi<<<(NBATCH * NSEQ) / 8, 256, 0, stream>>>(x, WI, s);

    // softmax + weighted row-sum partials, then reduce -> xbar
    k_softmax_xpart<<<dim3(NBATCH, 16), 1024, 0, stream>>>(x, s, xpart);
    k_reduce_xbar<<<(NBATCH * DIM) / 256, 256, 0, stream>>>(xpart, xbar);

    // cv = xbar @ WK
    k_cv<<<dim3(DIM / 128, NBATCH), 128, 0, stream>>>(xbar, WK, cvb);

    // fused: out = (relu(x@WV) * cv) @ WO
    const size_t lds = (size_t)2 * MT * XPAD * sizeof(__bf16);  // 264192 B
    k_fused<<<(NBATCH * NSEQ) / MT, 256, lds, stream>>>(x, wvt, wot, cvb, out);
}
